// SelfAttention_gate_72060961292402
// MI455X (gfx1250) — matrix operand order repaired and checked by reading, not compiled, not run on hardware
//
#include <hip/hip_runtime.h>
#include <hip/hip_bf16.h>

// Problem constants
#define NN_TOT 65536
#define TT     5
#define CC     256
#define HH     4
#define HD     64

// Tiling
#define NB       16            // batch items per workgroup
#define ROWS     (NB * TT)     // 80 rows = 5 M-tiles of 16
#define SEP      264           // padded bf16 LDS row stride (528B = 33*16B, breaks bank conflicts)
#define NTHREADS 512
#define NWAVES   (NTHREADS / 32)

typedef __bf16 bf16_t;
typedef bf16_t v8bf  __attribute__((ext_vector_type(8)));
typedef bf16_t v16bf __attribute__((ext_vector_type(16)));
typedef float  v8f   __attribute__((ext_vector_type(8)));
typedef float  v4f   __attribute__((ext_vector_type(4)));

__device__ __forceinline__ v16bf cat16(v8bf lo, v8bf hi) {
    return __builtin_shufflevector(lo, hi, 0,1,2,3,4,5,6,7,8,9,10,11,12,13,14,15);
}

// ---------------------------------------------------------------------------
// Prep: convert Wq/Wk/Wo (f32 row-major [K][N]) -> bf16 transposed [N][K] in ws.
// ---------------------------------------------------------------------------
__global__ void prep_weights_kernel(const float* __restrict__ Wq,
                                    const float* __restrict__ Wk,
                                    const float* __restrict__ Wo,
                                    bf16_t* __restrict__ Wt) {
    int idx = blockIdx.x * blockDim.x + threadIdx.x;
    if (idx >= 3 * CC * CC) return;
    int p  = idx >> 16;          // which weight matrix
    int e  = idx & 0xFFFF;
    int nn = e >> 8;             // output column (row of transposed)
    int kk = e & 0xFF;           // reduction index
    const float* W = (p == 0) ? Wq : (p == 1) ? Wk : Wo;
    Wt[idx] = (bf16_t)W[kk * CC + nn];
}

// ---------------------------------------------------------------------------
// Main fused kernel: 1 workgroup = 16 batch items (80 rows).
// ---------------------------------------------------------------------------
__global__ __launch_bounds__(NTHREADS)
void attn_gate_kernel(const float* __restrict__ emb,
                      const float* __restrict__ ac,
                      const float* __restrict__ bq,
                      const float* __restrict__ bk,
                      const float* __restrict__ bo,
                      const float* __restrict__ Wg,
                      const float* __restrict__ bg,
                      const bf16_t* __restrict__ Wt,
                      float* __restrict__ ctx_out,
                      float* __restrict__ attn_out) {
    __shared__ bf16_t sE[ROWS * SEP];          // embeddings (bf16)
    __shared__ bf16_t sQ[ROWS * SEP];          // Q projection; reused as context
    __shared__ bf16_t sK[ROWS * SEP];          // K projection
    __shared__ float  sAttn[NB][HH][TT][TT];   // attention weights
    __shared__ float  sGate[NB][TT];           // gate per (n, key)

    const int tid   = threadIdx.x;
    const int wave  = tid >> 5;
    const int lane  = tid & 31;
    const int half  = lane >> 4;     // 0: lanes 0-15, 1: lanes 16-31
    const int l15   = lane & 15;
    const int blk   = blockIdx.x;

    // ---- Phase A: stage embeddings (f32 global -> bf16 LDS), compute gate ----
    const float* eblk = emb + (size_t)blk * ROWS * CC;
    for (int i = tid; i < ROWS * CC / 4; i += NTHREADS) {
        v4f v = ((const v4f*)eblk)[i];
        int row = (i * 4) / CC, col = (i * 4) % CC;
        bf16_t* d = &sE[row * SEP + col];
        d[0] = (bf16_t)v[0]; d[1] = (bf16_t)v[1];
        d[2] = (bf16_t)v[2]; d[3] = (bf16_t)v[3];
    }
    if (tid < NB * TT) {
        int nl = tid / TT, j = tid % TT;
        const float* a = ac + ((size_t)blk * NB + nl) * TT;
        float g = bg[j];
        #pragma unroll
        for (int i = 0; i < TT; ++i) g += a[i] * Wg[i * TT + j];
        sGate[nl][j] = g;
    }
    __syncthreads();

    // ---- Phase B: Q and K projections via bf16 WMMA ----
    // 2 projections x 16 N-tiles x 5 M-tiles = 160 wave-jobs, 10 per wave.
    for (int job = wave; job < 2 * 16 * 5; job += NWAVES) {
        int proj  = job / 80;
        int rem   = job - proj * 80;
        int ntile = rem & 15;
        int mtile = rem >> 4;
        int m0 = mtile * 16, n0 = ntile * 16;
        const bf16_t* W = Wt + (size_t)proj * CC * CC;
        const int arow = m0 + l15;
        const int bcol = n0 + l15;
        v8f acc = {};
        #pragma unroll
        for (int k0 = 0; k0 < CC; k0 += 32) {
            // A fragment: lanes<16 hold K k0+0..7 / k0+16..23, lanes>=16 K k0+8..15 / k0+24..31
            int ka = k0 + (half ? 8 : 0);
            v8bf alo = *(const v8bf*)&sE[arow * SEP + ka];
            v8bf ahi = *(const v8bf*)&sE[arow * SEP + ka + 16];
            // B fragment: lanes<16 hold K k0+0..7 / k0+16..23, lanes>=16 K k0+8..15 / k0+24..31 (Wt is N-major)
            int kb = k0 + (half ? 8 : 0);
            v8bf blo = *(const v8bf*)&W[(size_t)bcol * CC + kb];
            v8bf bhi = *(const v8bf*)&W[(size_t)bcol * CC + kb + 16];
            acc = __builtin_amdgcn_wmma_f32_16x16x32_bf16(
                false, cat16(alo, ahi), false, cat16(blo, bhi),
                (short)0, acc, false, false);
        }
        float bias = (proj == 0 ? bq : bk)[bcol];
        bf16_t* dst = (proj == 0) ? sQ : sK;
        #pragma unroll
        for (int r = 0; r < 8; ++r) {
            int row = m0 + r + half * 8;           // D layout: VGPR r -> M = r (+8 for upper half)
            dst[row * SEP + bcol] = (bf16_t)(acc[r] + bias);
        }
    }
    __syncthreads();

    // ---- Phase C: scores, gate, top-3 mask, softmax; write attention output ----
    if (tid < NB * HH * TT) {
        int nl  = tid / (HH * TT);
        int rem = tid % (HH * TT);
        int h = rem / TT, q = rem % TT;
        int qrow = nl * TT + q;
        float s[TT];
        for (int k = 0; k < TT; ++k) {
            int krow = nl * TT + k;
            const bf16_t* qp = &sQ[qrow * SEP + h * HD];
            const bf16_t* kp = &sK[krow * SEP + h * HD];
            float d = 0.f;
            #pragma unroll
            for (int c8 = 0; c8 < HD; c8 += 8) {
                v8bf qv = *(const v8bf*)&qp[c8];
                v8bf kv = *(const v8bf*)&kp[c8];
                #pragma unroll
                for (int j = 0; j < 8; ++j) d += (float)qv[j] * (float)kv[j];
            }
            s[k] = d * 0.125f * sGate[nl][k];      // /sqrt(64) then gate
        }
        // top-3 of 5 with index tie-break (matches jax.lax.top_k + one_hot mask)
        bool keep[TT];
        #pragma unroll
        for (int i = 0; i < TT; ++i) {
            int rank = 0;
            #pragma unroll
            for (int j = 0; j < TT; ++j)
                if (s[j] > s[i] || (s[j] == s[i] && j < i)) rank++;
            keep[i] = (rank < 3);
        }
        float m = -__builtin_inff();
        #pragma unroll
        for (int i = 0; i < TT; ++i) if (keep[i] && s[i] > m) m = s[i];
        float e[TT], Z = 0.f;
        #pragma unroll
        for (int i = 0; i < TT; ++i) { e[i] = keep[i] ? __expf(s[i] - m) : 0.f; Z += e[i]; }
        float inv = 1.f / Z;
        size_t obase = ((((size_t)blk * NB + nl) * HH + h) * TT + q) * TT;
        #pragma unroll
        for (int k = 0; k < TT; ++k) {
            float aval = e[k] * inv;
            sAttn[nl][h][q][k] = aval;
            attn_out[obase + k] = aval;
        }
    }
    __syncthreads();

    // ---- Phase D: context = attn @ V (V = embeddings), bf16 into sQ ----
    for (int i = tid; i < ROWS * CC; i += NTHREADS) {
        int row = i / CC, c = i % CC;
        int nl = row / TT, q = row % TT, h = c / HD;
        float d = 0.f;
        #pragma unroll
        for (int k = 0; k < TT; ++k)
            d += sAttn[nl][h][q][k] * (float)sE[(nl * TT + k) * SEP + c];
        sQ[row * SEP + c] = (bf16_t)d;
    }
    __syncthreads();

    // ---- Phase E: out projection via bf16 WMMA, f32 store + bias ----
    for (int job = wave; job < 16 * 5; job += NWAVES) {
        int ntile = job & 15;
        int mtile = job >> 4;
        int m0 = mtile * 16, n0 = ntile * 16;
        const bf16_t* W = Wt + (size_t)2 * CC * CC;
        const int arow = m0 + l15;
        const int bcol = n0 + l15;
        v8f acc = {};
        #pragma unroll
        for (int k0 = 0; k0 < CC; k0 += 32) {
            int ka = k0 + (half ? 8 : 0);
            v8bf alo = *(const v8bf*)&sQ[arow * SEP + ka];
            v8bf ahi = *(const v8bf*)&sQ[arow * SEP + ka + 16];
            int kb = k0 + (half ? 8 : 0);
            v8bf blo = *(const v8bf*)&W[(size_t)bcol * CC + kb];
            v8bf bhi = *(const v8bf*)&W[(size_t)bcol * CC + kb + 16];
            acc = __builtin_amdgcn_wmma_f32_16x16x32_bf16(
                false, cat16(alo, ahi), false, cat16(blo, bhi),
                (short)0, acc, false, false);
        }
        float bias = bo[bcol];
        #pragma unroll
        for (int r = 0; r < 8; ++r) {
            int row = m0 + r + half * 8;
            ctx_out[((size_t)blk * ROWS + row) * CC + bcol] = acc[r] + bias;
        }
    }
}

// ---------------------------------------------------------------------------
extern "C" void kernel_launch(void* const* d_in, const int* in_sizes, int n_in,
                              void* d_out, int out_size, void* d_ws, size_t ws_size,
                              hipStream_t stream) {
    const float* emb = (const float*)d_in[0];
    const float* ac  = (const float*)d_in[1];
    const float* Wq  = (const float*)d_in[2];
    const float* bq  = (const float*)d_in[3];
    const float* Wk  = (const float*)d_in[4];
    const float* bk  = (const float*)d_in[5];
    const float* Wo  = (const float*)d_in[6];
    const float* bo  = (const float*)d_in[7];
    const float* Wg  = (const float*)d_in[8];
    const float* bg  = (const float*)d_in[9];

    bf16_t* Wt       = (bf16_t*)d_ws;                       // 3 * 256*256 bf16 = 384 KB
    float*  ctx      = (float*)d_out;                       // [N, T, C]
    float*  attn     = ctx + (size_t)NN_TOT * TT * CC;      // [N, H, T, T]

    prep_weights_kernel<<<(3 * CC * CC + 511) / 512, 512, 0, stream>>>(Wq, Wk, Wo, Wt);

    const int nblocks = NN_TOT / NB;                        // 4096
    attn_gate_kernel<<<nblocks, NTHREADS, 0, stream>>>(
        emb, ac, bq, bk, bo, Wg, bg, Wt, ctx, attn);
}
